// LeWin_46291157517053
// MI455X (gfx1250) — matrix operand order repaired and checked by reading, not compiled, not run on hardware
//
#include <hip/hip_runtime.h>

typedef _Float16 half_t;
typedef __attribute__((ext_vector_type(16))) _Float16 v16h;
typedef __attribute__((ext_vector_type(8)))  _Float16 v8h;
typedef __attribute__((ext_vector_type(8)))  float    v8f;

// ---------------------------------------------------------------------------
// WMMA helpers (wave32, 16x16x32 f16 -> f32), fragment layouts per ISA 7.12.2
// ---------------------------------------------------------------------------
__device__ __forceinline__ v8f wmma_f16(v16h a, v16h b, v8f c) {
  return __builtin_amdgcn_wmma_f32_16x16x32_f16(
      /*neg_a=*/false, a, /*neg_b=*/false, b,
      /*c_mod=*/(short)0, c, /*reuse_a=*/false, /*reuse_b=*/false);
}

// A: 16x32 f16, row-major, ld (halfs) multiple of 8, tile 16B-aligned:
// two global_load_b128 per lane.
__device__ __forceinline__ v16h frag_a_f16_vec(const half_t* tile, int ld) {
  const int lane = threadIdx.x & 31;
  const int m    = lane & 15;
  const int k0   = (lane >> 4) * 8;
  const half_t* p = tile + m * ld + k0;
  const v8h lo = *(const v8h*)(p);
  const v8h hi = *(const v8h*)(p + 16);
  v16h a;
#pragma unroll
  for (int j = 0; j < 8; ++j) { a[j] = lo[j]; a[j + 8] = hi[j]; }
  return a;
}

// A: 16x32 f16 from (possibly unaligned) LDS rows; scalar half loads.
__device__ __forceinline__ v16h frag_a_f16(const half_t* tile, int ld) {
  const int lane = threadIdx.x & 31;
  const int m    = lane & 15;
  const int k0   = (lane >> 4) * 8;
  const half_t* p = tile + m * ld + k0;
  v16h a;
#pragma unroll
  for (int j = 0; j < 8; ++j) a[j] = p[j];
#pragma unroll
  for (int j = 0; j < 8; ++j) a[j + 8] = p[16 + j];
  return a;
}

// A fragment from f32 LDS slots whose LOW 16 bits hold packed f16 values.
__device__ __forceinline__ v16h frag_a_lds_packed(const float* tile, int ld) {
  const int lane = threadIdx.x & 31;
  const int m    = lane & 15;
  const int k0   = (lane >> 4) * 8;
  const float* p = tile + m * ld + k0;
  v16h a;
#pragma unroll
  for (int j = 0; j < 8; ++j) a[j] = *(const half_t*)(p + j);
#pragma unroll
  for (int j = 0; j < 8; ++j) a[j + 8] = *(const half_t*)(p + 16 + j);
  return a;
}

// B: 32x16 f16, row-major [k][n] source with leading dim `ld` (halfs).
__device__ __forceinline__ v16h frag_b_f16(const half_t* tile, int ld) {
  const int lane = threadIdx.x & 31;
  const int n    = lane & 15;
  const int kb   = (lane >> 4) * 8;
  v16h b;
#pragma unroll
  for (int j = 0; j < 16; ++j) b[j] = tile[(kb + (j & 7) + ((j >> 3) << 4)) * ld + n];
  return b;
}

// Same but converting from an f32 row-major [k][n] weight matrix on the fly.
__device__ __forceinline__ v16h frag_b_f32src(const float* tile, int ld) {
  const int lane = threadIdx.x & 31;
  const int n    = lane & 15;
  const int kb   = (lane >> 4) * 8;
  v16h b;
#pragma unroll
  for (int j = 0; j < 16; ++j) b[j] = (half_t)tile[(kb + (j & 7) + ((j >> 3) << 4)) * ld + n];
  return b;
}

__device__ __forceinline__ float gelu_exact(float x) {
  return 0.5f * x * (1.0f + erff(x * 0.70710678118654752f));
}

// static config
#define CCH   32
#define NTOK  256
#define HEADS 8
#define ADIM  32
#define INNER 128
#define NWIN  512      /* B * NH * NH */
#define LDS_S 260      /* padded score row (floats) */

// ---------------------------------------------------------------------------
// K1: LN1 + cyclic shift + window partition + modulator + QKV GEMM
//     -> q (scaled, [W,H,N,A] f16), kT ([W,H,A,N] f16), v ([W,H,N,A] f16)
// ---------------------------------------------------------------------------
__global__ __launch_bounds__(256) void k_ln1_qkv(
    const float* __restrict__ x, const float* __restrict__ g1,
    const float* __restrict__ b1, const float* __restrict__ mod_w,
    const float* __restrict__ qkv_w, const float* __restrict__ qkv_b,
    half_t* __restrict__ q, half_t* __restrict__ kT, half_t* __restrict__ v) {
  __shared__ half_t sA[NTOK][36];
  const int w  = blockIdx.x;
  const int b  = w >> 6;
  const int wi = w & 63;
  const int t  = threadIdx.x;

  { // LN over C=32 for one token, at shifted source pixel
    const int rs = ((wi >> 3) << 4) + (t >> 4);
    const int cs = ((wi & 7) << 4) + (t & 15);
    const int r  = (rs + 8) & 127, c = (cs + 8) & 127;
    const float* px = x + (((size_t)b * 16384) + (size_t)r * 128 + c) * CCH;
    float buf[CCH];
    float mean = 0.f;
#pragma unroll
    for (int i = 0; i < CCH; ++i) { buf[i] = px[i]; mean += buf[i]; }
    mean *= (1.0f / CCH);
    float var = 0.f;
#pragma unroll
    for (int i = 0; i < CCH; ++i) { float d = buf[i] - mean; var += d * d; }
    var *= (1.0f / CCH);
    const float inv = rsqrtf(var + 1e-5f);
#pragma unroll
    for (int i = 0; i < CCH; ++i) {
      float val = (buf[i] - mean) * inv * g1[i] + b1[i] + mod_w[t * CCH + i];
      sA[t][i] = (half_t)val;
    }
  }
  __syncthreads();

  const int wid  = t >> 5;
  const int lane = t & 31;
  const int nn   = lane & 15;
  const int mb   = (lane >> 4) * 8;
#pragma unroll 1
  for (int jm = 0; jm < 2; ++jm) {
    const int tm = wid * 2 + jm;
    v16h a = frag_a_f16(&sA[tm * 16][0], 36);
#pragma unroll 1
    for (int tc = 0; tc < 48; ++tc) {
      v16h bb = frag_b_f32src(qkv_w + tc * 16, 768);
      v8f acc = {};
      acc = wmma_f16(a, bb, acc);
      // tile-uniform routing: a 16-col tile never straddles a q/k/v boundary
      const int three = tc >> 4;               // uniform: 0=q 1=k 2=v
      const int hh    = (tc & 15) >> 1;        // uniform head
      const int aa    = ((tc & 1) << 4) + nn;  // head-dim column
      const float bias = qkv_b[tc * 16 + nn];
      const size_t whb = (size_t)w * HEADS + hh;
      if (three == 0) {
#pragma unroll
        for (int rr = 0; rr < 8; ++rr) {
          const int token = tm * 16 + mb + rr;
          q[(whb * NTOK + token) * ADIM + aa] =
              (half_t)((acc[rr] + bias) * 0.17677669529663689f);
        }
      } else if (three == 1) {
#pragma unroll
        for (int rr = 0; rr < 8; ++rr) {
          const int token = tm * 16 + mb + rr;
          kT[(whb * ADIM + aa) * NTOK + token] = (half_t)(acc[rr] + bias);
        }
      } else {
#pragma unroll
        for (int rr = 0; rr < 8; ++rr) {
          const int token = tm * 16 + mb + rr;
          v[(whb * NTOK + token) * ADIM + aa] = (half_t)(acc[rr] + bias);
        }
      }
    }
  }
}

// ---------------------------------------------------------------------------
// K2: per (window, head) attention: S = q kT (+rpb +shift-mask), softmax,
//     O = P v  -> o [W, N, H*A] f16.  LDS: 256x260 f32 scores + code table.
// ---------------------------------------------------------------------------
__global__ __launch_bounds__(256) void k_attn(
    const half_t* __restrict__ q, const half_t* __restrict__ kT,
    const half_t* __restrict__ v, const float* __restrict__ rpbt,
    half_t* __restrict__ o) {
  extern __shared__ char smem[];
  float* sS   = (float*)smem;                     // [256][260]
  int* sCode  = (int*)(smem + NTOK * LDS_S * 4);  // [256]

  const int wh = blockIdx.x;
  const int w  = wh >> 3;
  const int hh = wh & 7;
  const int wi = w & 63;
  const int t  = threadIdx.x;

  { // shift-mask region code for token t of this window
    const int rs = ((wi >> 3) << 4) + (t >> 4);
    const int cs = ((wi & 7) << 4) + (t & 15);
    const int cr = rs < 112 ? 0 : (rs < 120 ? 1 : 2);
    const int cc = cs < 112 ? 0 : (cs < 120 ? 1 : 2);
    sCode[t] = cr * 3 + cc;
  }
  __syncthreads();

  const half_t* qh = q  + (size_t)wh * NTOK * ADIM;
  const half_t* kh = kT + (size_t)wh * ADIM * NTOK;
  const half_t* vh = v  + (size_t)wh * NTOK * ADIM;

  const int wid  = t >> 5;
  const int lane = t & 31;
  const int nn   = lane & 15;
  const int mb   = (lane >> 4) * 8;

  // ---- S = q @ kT, bias + mask fused at writeback ----
#pragma unroll 1
  for (int jm = 0; jm < 2; ++jm) {
    const int tm = wid * 2 + jm;
    v16h a = frag_a_f16_vec(qh + tm * 16 * ADIM, ADIM);
#pragma unroll 1
    for (int tn = 0; tn < 16; ++tn) {
      v16h bb = frag_b_f16(kh + tn * 16, NTOK);
      v8f acc = {};
      acc = wmma_f16(a, bb, acc);
#pragma unroll
      for (int rr = 0; rr < 8; ++rr) {
        const int ng = tm * 16 + mb + rr;   // query token
        const int mg = tn * 16 + nn;        // key token
        const int d0 = (ng >> 4) - (mg >> 4);
        const int d1 = (ng & 15) - (mg & 15);
        const int idx = (d0 + 15) * 31 + (d1 + 15);
        float val = acc[rr] + rpbt[idx * HEADS + hh];
        if (sCode[ng] != sCode[mg]) val -= 100.0f;
        sS[ng * LDS_S + mg] = val;
      }
    }
  }
  __syncthreads();

  // ---- softmax, one thread per row; repack f16 into low half of f32 slot ----
  {
    float* row = sS + t * LDS_S;
    float mx = -1e30f;
    for (int m = 0; m < NTOK; ++m) mx = fmaxf(mx, row[m]);
    float s = 0.f;
    for (int m = 0; m < NTOK; ++m) s += __expf(row[m] - mx);
    const float inv = 1.0f / s;
    for (int m = 0; m < NTOK; ++m) {
      half_t pv = (half_t)(__expf(row[m] - mx) * inv);
      *(half_t*)(row + m) = pv;   // own row only: race-free
    }
  }
  __syncthreads();

  // ---- O = P @ v (K = 512 -> 16 WMMA steps) ----
#pragma unroll 1
  for (int i = 0; i < 4; ++i) {
    const int tm = wid * 2 + (i & 1);
    const int ta = i >> 1;
    v8f acc = {};
#pragma unroll 1
    for (int kk = 0; kk < 16; ++kk) {
      v16h a  = frag_a_lds_packed(sS + tm * 16 * LDS_S + kk * 32, LDS_S);
      v16h bb = frag_b_f16(vh + (kk * 32) * ADIM + ta * 16, ADIM);
      acc = wmma_f16(a, bb, acc);
    }
#pragma unroll
    for (int rr = 0; rr < 8; ++rr) {
      const int token = tm * 16 + mb + rr;
      o[((size_t)w * NTOK + token) * (HEADS * ADIM) + hh * ADIM + ta * 16 + nn] =
          (half_t)acc[rr];
    }
  }
}

// ---------------------------------------------------------------------------
// K3: o @ rev_w + rev_b, window-reverse + un-shift, + shortcut -> x_res (f32)
// ---------------------------------------------------------------------------
__global__ __launch_bounds__(256) void k_rev(
    const half_t* __restrict__ o, const float* __restrict__ rev_w,
    const float* __restrict__ rev_b, const float* __restrict__ x,
    float* __restrict__ xres) {
  const int w  = blockIdx.x;
  const int b  = w >> 6;
  const int wi = w & 63;
  const int t  = threadIdx.x;
  const int wid  = t >> 5;
  const int lane = t & 31;
  const int nn   = lane & 15;
  const int mb   = (lane >> 4) * 8;
  const half_t* oh = o + (size_t)w * NTOK * (HEADS * ADIM);

#pragma unroll 1
  for (int i = 0; i < 4; ++i) {
    const int tm = wid * 2 + (i & 1);
    const int tc = i >> 1;
    v8f acc = {};
#pragma unroll 1
    for (int kk = 0; kk < 8; ++kk) {   // K = 256
      v16h a  = frag_a_f16_vec(oh + tm * 16 * 256 + kk * 32, 256);
      v16h bb = frag_b_f32src(rev_w + kk * 32 * CCH + tc * 16, CCH);
      acc = wmma_f16(a, bb, acc);
    }
    const int ch = tc * 16 + nn;
    const float bias = rev_b[ch];
#pragma unroll
    for (int rr = 0; rr < 8; ++rr) {
      const int token = tm * 16 + mb + rr;
      const int rs = ((wi >> 3) << 4) + (token >> 4);
      const int cs = ((wi & 7) << 4) + (token & 15);
      const int r  = (rs + 8) & 127, c = (cs + 8) & 127;
      const size_t pix = (size_t)b * 16384 + (size_t)r * 128 + c;
      xres[pix * CCH + ch] = x[pix * CCH + ch] + acc[rr] + bias;
    }
  }
}

// ---------------------------------------------------------------------------
// K4: LN2 + lp1 GEMM + GELU -> y1 ([B*16384, 128] f16)
// ---------------------------------------------------------------------------
__global__ __launch_bounds__(256) void k_ln2_lp1(
    const float* __restrict__ xres, const float* __restrict__ g2,
    const float* __restrict__ b2, const float* __restrict__ lp1_w,
    const float* __restrict__ lp1_b, half_t* __restrict__ y1) {
  __shared__ half_t sA[NTOK][36];
  const int p0 = blockIdx.x * NTOK;
  const int t  = threadIdx.x;
  {
    const float* px = xres + (size_t)(p0 + t) * CCH;
    float buf[CCH];
    float mean = 0.f;
#pragma unroll
    for (int i = 0; i < CCH; ++i) { buf[i] = px[i]; mean += buf[i]; }
    mean *= (1.0f / CCH);
    float var = 0.f;
#pragma unroll
    for (int i = 0; i < CCH; ++i) { float d = buf[i] - mean; var += d * d; }
    var *= (1.0f / CCH);
    const float inv = rsqrtf(var + 1e-5f);
#pragma unroll
    for (int i = 0; i < CCH; ++i)
      sA[t][i] = (half_t)((buf[i] - mean) * inv * g2[i] + b2[i]);
  }
  __syncthreads();

  const int wid  = t >> 5;
  const int lane = t & 31;
  const int nn   = lane & 15;
  const int mb   = (lane >> 4) * 8;
#pragma unroll 1
  for (int jm = 0; jm < 2; ++jm) {
    const int tm = wid * 2 + jm;
    v16h a = frag_a_f16(&sA[tm * 16][0], 36);
#pragma unroll 1
    for (int tc = 0; tc < 8; ++tc) {
      v16h bb = frag_b_f32src(lp1_w + tc * 16, INNER);
      v8f acc = {};
      acc = wmma_f16(a, bb, acc);
      const int col = tc * 16 + nn;
      const float bias = lp1_b[col];
#pragma unroll
      for (int rr = 0; rr < 8; ++rr) {
        const int p = p0 + tm * 16 + mb + rr;
        y1[(size_t)p * INNER + col] = (half_t)gelu_exact(acc[rr] + bias);
      }
    }
  }
}

// ---------------------------------------------------------------------------
// K5: 3x3 SAME conv 128->128 as 9x4 accumulated WMMA K-steps, + bias + GELU
// ---------------------------------------------------------------------------
__global__ __launch_bounds__(256) void k_conv(
    const half_t* __restrict__ y1, const float* __restrict__ conv_w,
    const float* __restrict__ conv_b, half_t* __restrict__ y2) {
  const int blk  = blockIdx.x;
  const int b    = blk >> 6;
  const int tile = blk & 63;
  const int tr   = tile >> 3;
  const int tcol = tile & 7;
  const int t    = threadIdx.x;
  const int wid  = t >> 5;
  const int lane = t & 31;
  const int nn   = lane & 15;
  const int mb   = (lane >> 4) * 8;
  const int k0   = (lane >> 4) * 8;

#pragma unroll 1
  for (int i = 0; i < 16; ++i) {
    const int tm = wid * 2 + (i >> 3);
    const int tc = i & 7;
    v8f acc = {};
    // this lane's A-fragment source pixel (row of the 16x32 A tile)
    const int mloc = tm * 16 + (lane & 15);
    const int py   = tr * 16 + (mloc >> 4);
    const int px   = tcol * 16 + (mloc & 15);
#pragma unroll 1
    for (int ky = 0; ky < 3; ++ky) {
#pragma unroll 1
      for (int kx = 0; kx < 3; ++kx) {
        const int sy = py + ky - 1;
        const int sx = px + kx - 1;
        const bool ok = (sy >= 0) & (sy < 128) & (sx >= 0) & (sx < 128);
        const half_t* prow =
            y1 + (((size_t)b * 16384 + (size_t)(ok ? sy : 0) * 128 +
                   (ok ? sx : 0)) * INNER);
#pragma unroll 1
        for (int kk = 0; kk < 4; ++kk) {
          v16h a = {};
          if (ok) {
            const half_t* p = prow + kk * 32 + k0;
            const v8h lo = *(const v8h*)(p);
            const v8h hi = *(const v8h*)(p + 16);
#pragma unroll
            for (int j = 0; j < 8; ++j) { a[j] = lo[j]; a[j + 8] = hi[j]; }
          }
          v16h bb = frag_b_f32src(
              conv_w + ((size_t)(ky * 3 + kx) * INNER + kk * 32) * INNER + tc * 16,
              INNER);
          acc = wmma_f16(a, bb, acc);
        }
      }
    }
    const int oc = tc * 16 + nn;
    const float bias = conv_b[oc];
#pragma unroll
    for (int rr = 0; rr < 8; ++rr) {
      const int m  = tm * 16 + mb + rr;
      const int oy = tr * 16 + (m >> 4);
      const int ox = tcol * 16 + (m & 15);
      y2[(((size_t)b * 16384 + (size_t)oy * 128 + ox) * INNER) + oc] =
          (half_t)gelu_exact(acc[rr] + bias);
    }
  }
}

// ---------------------------------------------------------------------------
// K6: y2 @ lp2_w + b, GELU, + x_res -> out (f32)
// ---------------------------------------------------------------------------
__global__ __launch_bounds__(256) void k_lp2(
    const half_t* __restrict__ y2, const float* __restrict__ lp2_w,
    const float* __restrict__ lp2_b, const float* __restrict__ xres,
    float* __restrict__ out) {
  const int p0 = blockIdx.x * NTOK;
  const int t  = threadIdx.x;
  const int wid  = t >> 5;
  const int lane = t & 31;
  const int nn   = lane & 15;
  const int mb   = (lane >> 4) * 8;

#pragma unroll 1
  for (int i = 0; i < 4; ++i) {
    const int tm = wid * 2 + (i & 1);
    const int tc = i >> 1;
    v8f acc = {};
#pragma unroll 1
    for (int kk = 0; kk < 4; ++kk) {   // K = 128
      v16h a  = frag_a_f16_vec(y2 + (size_t)(p0 + tm * 16) * INNER + kk * 32,
                               INNER);
      v16h bb = frag_b_f32src(lp2_w + kk * 32 * CCH + tc * 16, CCH);
      acc = wmma_f16(a, bb, acc);
    }
    const int ch = tc * 16 + nn;
    const float bias = lp2_b[ch];
#pragma unroll
    for (int rr = 0; rr < 8; ++rr) {
      const size_t p = (size_t)p0 + tm * 16 + mb + rr;
      out[p * CCH + ch] = gelu_exact(acc[rr] + bias) + xres[p * CCH + ch];
    }
  }
}

// ---------------------------------------------------------------------------
extern "C" void kernel_launch(void* const* d_in, const int* in_sizes, int n_in,
                              void* d_out, int out_size, void* d_ws,
                              size_t ws_size, hipStream_t stream) {
  (void)in_sizes; (void)n_in; (void)out_size; (void)ws_size;
  const float* x       = (const float*)d_in[0];
  const float* norm1_g = (const float*)d_in[1];
  const float* norm1_b = (const float*)d_in[2];
  const float* mod_w   = (const float*)d_in[3];
  const float* qkv_w   = (const float*)d_in[4];
  const float* qkv_b   = (const float*)d_in[5];
  const float* rpbt    = (const float*)d_in[6];
  const float* rev_w   = (const float*)d_in[7];
  const float* rev_b   = (const float*)d_in[8];
  const float* norm2_g = (const float*)d_in[9];
  const float* norm2_b = (const float*)d_in[10];
  const float* lp1_w   = (const float*)d_in[11];
  const float* lp1_b   = (const float*)d_in[12];
  const float* conv_w  = (const float*)d_in[13];
  const float* conv_b  = (const float*)d_in[14];
  const float* lp2_w   = (const float*)d_in[15];
  const float* lp2_b   = (const float*)d_in[16];
  float* out = (float*)d_out;

  char* ws = (char*)d_ws;
  half_t* q    = (half_t*)(ws);                          // 64 MB
  half_t* kT   = (half_t*)(ws + ((size_t)64 << 20));     // 64 MB
  half_t* v    = (half_t*)(ws + ((size_t)128 << 20));    // 64 MB
  half_t* o    = (half_t*)(ws + ((size_t)192 << 20));    // 64 MB
  float*  xres = (float*)(ws + ((size_t)256 << 20));     // 16 MB
  half_t* y1   = (half_t*)(ws);                          // reuse q region
  half_t* y2   = (half_t*)(ws + ((size_t)64 << 20));     // reuse kT region

  const size_t smem2 = (size_t)NTOK * LDS_S * 4 + NTOK * 4; // 267,264 B < 320 KB

  k_ln1_qkv<<<NWIN, 256, 0, stream>>>(x, norm1_g, norm1_b, mod_w, qkv_w, qkv_b,
                                      q, kT, v);
  k_attn<<<NWIN * HEADS, 256, smem2, stream>>>(q, kT, v, rpbt, o);
  k_rev<<<NWIN, 256, 0, stream>>>(o, rev_w, rev_b, x, xres);
  k_ln2_lp1<<<512, 256, 0, stream>>>(xres, norm2_g, norm2_b, lp1_w, lp1_b, y1);
  k_conv<<<512, 256, 0, stream>>>(y1, conv_w, conv_b, y2);
  k_lp2<<<512, 256, 0, stream>>>(y2, lp2_w, lp2_b, xres, out);
}
